// LocalAttentionBlock_50010599194638
// MI455X (gfx1250) — matrix operand order repaired and checked by reading, not compiled, not run on hardware
//
#include <hip/hip_runtime.h>

typedef __bf16 bf16;
typedef __attribute__((ext_vector_type(16))) __bf16 v16bf;
typedef __attribute__((ext_vector_type(8)))  float  v8f;

#define DEV __device__ __forceinline__

// ---------------- problem constants ----------------
constexpr int B_   = 32;
constexpr int C_   = 320;   // out channels
constexpr int H_   = 64;
constexpr int W_   = 64;
constexpr int CTX  = 1024;  // ctx channels
constexpr int S_   = 256;   // ctx spatial (16x16)
constexpr int ROI  = 9;
constexpr int P_   = 81;    // roi spatial
constexpr int PP   = 96;    // padded roi spatial (6 WMMA tiles)
constexpr int KIN  = 322;   // 320 + 2 indicator
constexpr int KINP = 352;   // padded to 11*32
constexpr int MT   = C_ / 16;  // 20 M tiles
constexpr int MG   = MT / 4;   // 5 groups of 4 M tiles

constexpr size_t OUT0_ELEMS = (size_t)B_ * C_ * H_ * W_;  // 41,943,040 (out)
// attn output follows at OUT0_ELEMS: [B][81][256] f32

// ---------------- workspace layout (bytes) ----------------
constexpr size_t OFF_CTXT  = 0;                                  // bf16 [B][256][1024] raw ctx, p-major
constexpr size_t OFF_KMAPT = OFF_CTXT  + (size_t)B_*S_*CTX*2;    // bf16 [B][256][1024] groupnormed ctx
constexpr size_t OFF_WCTX  = OFF_KMAPT + (size_t)B_*S_*CTX*2;    // bf16 [320][1024]
constexpr size_t OFF_WOUT  = OFF_WCTX  + (size_t)C_*CTX*2;       // bf16 [9][320][1024] conv taps
constexpr size_t OFF_WIN   = OFF_WOUT  + (size_t)9*C_*CTX*2;     // bf16 [320][352]
constexpr size_t OFF_KT    = OFF_WIN   + (size_t)C_*KINP*2;      // bf16 [B][256][320]  k (pixel-major)
constexpr size_t OFF_VO    = OFF_KT    + (size_t)B_*S_*C_*2;     // bf16 [B][320][256]  v (channel-major)
constexpr size_t OFF_XCT   = OFF_VO    + (size_t)B_*C_*S_*2;     // bf16 [B][96][352]   x_cat (pixel-major)
constexpr size_t OFF_QT    = OFF_XCT   + (size_t)B_*PP*KINP*2;   // bf16 [B][96][320]   q
constexpr size_t OFF_SIM   = OFF_QT    + (size_t)B_*PP*C_*2;     // f32  [B][96][256]
constexpr size_t OFF_ATT   = OFF_SIM   + (size_t)B_*PP*S_*4;     // bf16 [B][96][256]
constexpr size_t OFF_XO    = OFF_ATT   + (size_t)B_*PP*S_*2;     // f32  [B][320][81]

// ---------------- WMMA fragment loaders (wave32, 16x16x32 bf16) ----------------
// A (16x32, row-major, lda): lane m=lane&15; lanes>=16 shift K by 16.
// elements 0..15 -> K = hi+0..15
DEV v16bf frag_a(const bf16* __restrict__ A, int lda, int lane) {
  int m  = lane & 15;
  int hi = (lane >> 4) << 4;          // 0 or 16
  const bf16* r = A + (size_t)m * lda + hi;
  v16bf a;
#pragma unroll
  for (int j = 0; j < 8; ++j)  a[j] = r[j];
#pragma unroll
  for (int j = 8; j < 16; ++j) a[j] = r[j];
  return a;
}
// B (32x16) from column-major storage: colp points at this lane's column (K contiguous).
// lanes 0..15 take K 0..15, lanes 16..31 take K 16..31.
DEV v16bf frag_b_col(const bf16* __restrict__ colp, int lane, bool valid) {
  int kb = (lane >> 4) << 4;          // 0 or 16
  v16bf v;
#pragma unroll
  for (int j = 0; j < 16; ++j) v[j] = valid ? colp[kb + j] : (bf16)0.0f;
  return v;
}
DEV v8f wmma_bf16(v16bf a, v16bf b, v8f c) {
  return __builtin_amdgcn_wmma_f32_16x16x32_bf16(false, a, false, b, (short)0, c, false, false);
}

// ---------------- K0: group-norm ctx_map, emit bf16 raw + normalized (p-major) ----------------
__global__ void k_ctx_norm(const float* __restrict__ context,
                           const float* __restrict__ cn_g, const float* __restrict__ cn_b,
                           bf16* __restrict__ ctxT, bf16* __restrict__ kmapT) {
  __shared__ float buf[8192];
  __shared__ float ss[256], sq[256];
  int b  = blockIdx.x >> 5;
  int g  = blockIdx.x & 31;
  int c0 = g * 32;                                   // 32 channels per group
  const float* src = context + (size_t)b * S_ * CTX; // context[b][p][c]
  float s = 0.f, q = 0.f;
  for (int e = threadIdx.x; e < 8192; e += 256) {
    int p = e >> 5, c = c0 + (e & 31);
    float v = src[(size_t)p * CTX + c];
    buf[e] = v; s += v; q += v * v;
  }
  ss[threadIdx.x] = s; sq[threadIdx.x] = q; __syncthreads();
  for (int o = 128; o > 0; o >>= 1) {
    if (threadIdx.x < o) { ss[threadIdx.x] += ss[threadIdx.x + o]; sq[threadIdx.x] += sq[threadIdx.x + o]; }
    __syncthreads();
  }
  float mean = ss[0] * (1.f / 8192.f);
  float var  = sq[0] * (1.f / 8192.f) - mean * mean;
  float rstd = rsqrtf(var + 1e-6f);
  bf16* oc = ctxT  + (size_t)b * S_ * CTX;
  bf16* ok = kmapT + (size_t)b * S_ * CTX;
  for (int e = threadIdx.x; e < 8192; e += 256) {
    int p = e >> 5, c = c0 + (e & 31);
    float v = buf[e];
    oc[(size_t)p * CTX + c] = (bf16)v;
    ok[(size_t)p * CTX + c] = (bf16)((v - mean) * rstd * cn_g[c] + cn_b[c]);
  }
}

// ---------------- K1: cast/repack weights to bf16 ----------------
__global__ void k_pack_w(const float* __restrict__ w_ctx, const float* __restrict__ w_out,
                         const float* __restrict__ w_in,
                         bf16* __restrict__ wctx, bf16* __restrict__ wout, bf16* __restrict__ win) {
  int idx = blockIdx.x * 256 + threadIdx.x;
  constexpr int N0 = C_ * CTX;        // 327680
  constexpr int N1 = 9 * C_ * CTX;    // 2949120
  constexpr int N2 = C_ * KINP;       // 112640
  if (idx < N0) { wctx[idx] = (bf16)w_ctx[idx]; return; }
  idx -= N0;
  if (idx < N1) {                     // wout[t][o][i] = w_out[o][i][t]
    int t = idx / (C_ * CTX);
    int r = idx % (C_ * CTX);
    int o = r / CTX, i = r % CTX;
    wout[idx] = (bf16)w_out[(size_t)o * CTX * 9 + (size_t)i * 9 + t];
    return;
  }
  idx -= N1;
  if (idx < N2) {                     // win[o][i], zero-padded K 322->352
    int o = idx / KINP, i = idx % KINP;
    win[idx] = (i < KIN) ? (bf16)w_in[o * KIN + i] : (bf16)0.0f;
  }
}

// ---------------- K2: k = w_ctx . kmap + b_ctx  -> kT[b][pixel][chan] bf16 ----------------
// register-blocked 4 M-tiles x 1 N-tile : B fragment reused by 4 wmma
__global__ void k_gemm_k(const bf16* __restrict__ wctx, const bf16* __restrict__ kmapT,
                         const float* __restrict__ b_ctx, bf16* __restrict__ kT) {
  int tile = blockIdx.x * 8 + (threadIdx.x >> 5);
  int lane = threadIdx.x & 31;
  if (tile >= B_ * MG * 16) return;
  int b  = tile / (MG * 16);
  int r  = tile % (MG * 16);
  int mg = r / 16, nt = r % 16;
  const bf16* A0 = wctx + (size_t)(mg * 64) * CTX;
  int n = lane & 15;
  const bf16* colp = kmapT + (size_t)b * S_ * CTX + (size_t)(nt * 16 + n) * CTX;
  v8f acc0 = {}, acc1 = {}, acc2 = {}, acc3 = {};
  for (int kt = 0; kt < CTX / 32; ++kt) {
    __builtin_prefetch(colp + (kt + 1) * 32, 0, 0);
    v16bf bb = frag_b_col(colp + kt * 32, lane, true);
    v16bf a0 = frag_a(A0 + (size_t)0  * CTX + kt * 32, CTX, lane);
    v16bf a1 = frag_a(A0 + (size_t)16 * CTX + kt * 32, CTX, lane);
    v16bf a2 = frag_a(A0 + (size_t)32 * CTX + kt * 32, CTX, lane);
    v16bf a3 = frag_a(A0 + (size_t)48 * CTX + kt * 32, CTX, lane);
    acc0 = wmma_bf16(a0, bb, acc0);
    acc1 = wmma_bf16(a1, bb, acc1);
    acc2 = wmma_bf16(a2, bb, acc2);
    acc3 = wmma_bf16(a3, bb, acc3);
  }
  int mb = (lane >> 4) << 3;
  bf16* out = kT + (size_t)b * S_ * C_ + (size_t)(nt * 16 + n) * C_ + mg * 64 + mb;
  const float* bias = b_ctx + mg * 64 + mb;
#pragma unroll
  for (int jj = 0; jj < 8; ++jj) out[jj]      = (bf16)(acc0[jj] + bias[jj]);
#pragma unroll
  for (int jj = 0; jj < 8; ++jj) out[16 + jj] = (bf16)(acc1[jj] + bias[16 + jj]);
#pragma unroll
  for (int jj = 0; jj < 8; ++jj) out[32 + jj] = (bf16)(acc2[jj] + bias[32 + jj]);
#pragma unroll
  for (int jj = 0; jj < 8; ++jj) out[48 + jj] = (bf16)(acc3[jj] + bias[48 + jj]);
}

// ---------------- K3: v = conv3x3(ctx) + b_out -> vO[b][chan][pixel] bf16 ----------------
// register-blocked 4 M-tiles x 1 output row : B (shifted ctx column) reused by 4 wmma
__global__ void k_conv_v(const bf16* __restrict__ wout, const bf16* __restrict__ ctxT,
                         const float* __restrict__ b_out, bf16* __restrict__ vO) {
  int tile = blockIdx.x * 8 + (threadIdx.x >> 5);
  int lane = threadIdx.x & 31;
  if (tile >= B_ * MG * 16) return;
  int b  = tile / (MG * 16);
  int r  = tile % (MG * 16);
  int mg = r / 16, y = r % 16;       // one output image row per wave (W=16 == N tile)
  int x  = lane & 15;
  const bf16* ctxb = ctxT + (size_t)b * S_ * CTX;
  v8f acc0 = {}, acc1 = {}, acc2 = {}, acc3 = {};
#pragma unroll
  for (int t = 0; t < 9; ++t) {
    int dy = t / 3 - 1, dx = t % 3 - 1;
    int yi = y + dy;
    if (yi < 0 || yi > 15) continue;               // wave-uniform skip, EXEC stays full
    int  xi = x + dx;
    bool cv = (xi >= 0) && (xi <= 15);             // per-lane column mask -> zero fragment
    const bf16* A0   = wout + (size_t)t * C_ * CTX + (size_t)(mg * 64) * CTX;
    const bf16* colp = ctxb + (size_t)(yi * 16 + (cv ? xi : 0)) * CTX;
    for (int kt = 0; kt < CTX / 32; ++kt) {
      __builtin_prefetch(colp + (kt + 1) * 32, 0, 0);
      v16bf bb = frag_b_col(colp + kt * 32, lane, cv);
      v16bf a0 = frag_a(A0 + (size_t)0  * CTX + kt * 32, CTX, lane);
      v16bf a1 = frag_a(A0 + (size_t)16 * CTX + kt * 32, CTX, lane);
      v16bf a2 = frag_a(A0 + (size_t)32 * CTX + kt * 32, CTX, lane);
      v16bf a3 = frag_a(A0 + (size_t)48 * CTX + kt * 32, CTX, lane);
      acc0 = wmma_bf16(a0, bb, acc0);
      acc1 = wmma_bf16(a1, bb, acc1);
      acc2 = wmma_bf16(a2, bb, acc2);
      acc3 = wmma_bf16(a3, bb, acc3);
    }
  }
  int mb = (lane >> 4) << 3;
  int p  = y * 16 + x;
  int m0 = mg * 64 + mb;
  bf16* out = vO + (size_t)b * C_ * S_ + (size_t)m0 * S_ + p;
#pragma unroll
  for (int jj = 0; jj < 8; ++jj) out[(size_t)(jj)      * S_] = (bf16)(acc0[jj] + b_out[m0 + jj]);
#pragma unroll
  for (int jj = 0; jj < 8; ++jj) out[(size_t)(16 + jj) * S_] = (bf16)(acc1[jj] + b_out[m0 + 16 + jj]);
#pragma unroll
  for (int jj = 0; jj < 8; ++jj) out[(size_t)(32 + jj) * S_] = (bf16)(acc2[jj] + b_out[m0 + 32 + jj]);
#pragma unroll
  for (int jj = 0; jj < 8; ++jj) out[(size_t)(48 + jj) * S_] = (bf16)(acc3[jj] + b_out[m0 + 48 + jj]);
}

// ---------------- K4: ROI-align + group-norm -> xcT[b][pixel][chan] bf16 ----------------
__global__ void k_roi_gn(const float* __restrict__ gx, const float* __restrict__ bbox,
                         const float* __restrict__ ln_g, const float* __restrict__ ln_b,
                         bf16* __restrict__ xcT) {
  __shared__ float buf[810];
  __shared__ float ss[256], sq[256];
  int b  = blockIdx.x >> 5;
  int g  = blockIdx.x & 31;
  int c0 = g * 10;                       // 10 channels per group (C=320, 32 groups)
  float x1 = bbox[b * 4 + 0], y1 = bbox[b * 4 + 1];
  float x2 = bbox[b * 4 + 2], y2 = bbox[b * 4 + 3];
  float rw = fmaxf(x2 - x1, 1.0f), rh = fmaxf(y2 - y1, 1.0f);
  const float* img = gx + (size_t)b * C_ * H_ * W_;
  float s = 0.f, q = 0.f;
  for (int e = threadIdx.x; e < 810; e += 256) {
    int c = c0 + e / 81, p = e % 81;
    int py = p / 9, px = p % 9;
    float xv = x1 + (px + 0.5f) * (rw * (1.0f / 9.0f));
    float yv = y1 + (py + 0.5f) * (rh * (1.0f / 9.0f));
    xv = fminf(fmaxf(xv, 0.f), (float)(W_ - 1));
    yv = fminf(fmaxf(yv, 0.f), (float)(H_ - 1));
    int x0 = (int)floorf(xv), y0 = (int)floorf(yv);
    int x1i = min(x0 + 1, W_ - 1), y1i = min(y0 + 1, H_ - 1);
    float lx = xv - (float)x0, ly = yv - (float)y0;
    const float* ch = img + (size_t)c * H_ * W_;
    float g00 = ch[y0 * W_ + x0],  g01 = ch[y0 * W_ + x1i];
    float g10 = ch[y1i * W_ + x0], g11 = ch[y1i * W_ + x1i];
    float v = g00 * (1 - ly) * (1 - lx) + g01 * (1 - ly) * lx
            + g10 * ly * (1 - lx)       + g11 * ly * lx;
    buf[e] = v; s += v; q += v * v;
  }
  ss[threadIdx.x] = s; sq[threadIdx.x] = q; __syncthreads();
  for (int o = 128; o > 0; o >>= 1) {
    if (threadIdx.x < o) { ss[threadIdx.x] += ss[threadIdx.x + o]; sq[threadIdx.x] += sq[threadIdx.x + o]; }
    __syncthreads();
  }
  float mean = ss[0] * (1.f / 810.f);
  float var  = sq[0] * (1.f / 810.f) - mean * mean;
  float rstd = rsqrtf(var + 1e-6f);
  for (int e = threadIdx.x; e < 810; e += 256) {
    int c = c0 + e / 81, p = e % 81;
    float v = (buf[e] - mean) * rstd * ln_g[c] + ln_b[c];
    xcT[(size_t)b * PP * KINP + (size_t)p * KINP + c] = (bf16)v;
  }
}

// ---------------- K4b: indicator rows + zero padding of x_cat ----------------
__global__ void k_xcat_pad(const float* __restrict__ ind, bf16* __restrict__ xcT) {
  int b = blockIdx.x;
  bf16 i0 = (bf16)ind[b * 2 + 0], i1 = (bf16)ind[b * 2 + 1];
  bf16* base = xcT + (size_t)b * PP * KINP;
  for (int e = threadIdx.x; e < PP * 32; e += 256) {      // channels 320..351, all pixels
    int p = e / 32, c = 320 + (e % 32);
    base[(size_t)p * KINP + c] = (c == 320) ? i0 : (c == 321) ? i1 : (bf16)0.0f;
  }
  for (int e = threadIdx.x; e < 15 * 320; e += 256) {     // pixels 81..95, channels 0..319
    int p = 81 + e / 320, c = e % 320;
    base[(size_t)p * KINP + c] = (bf16)0.0f;
  }
}

// ---------------- K5: q = w_in . x_cat + b_in -> qT[b][pixel][chan] bf16 ----------------
// register-blocked 4 M-tiles x 1 N-tile
__global__ void k_gemm_q(const bf16* __restrict__ win, const bf16* __restrict__ xcT,
                         const float* __restrict__ b_in, bf16* __restrict__ qT) {
  int tile = blockIdx.x * 8 + (threadIdx.x >> 5);
  int lane = threadIdx.x & 31;
  if (tile >= B_ * MG * 6) return;
  int b  = tile / (MG * 6);
  int r  = tile % (MG * 6);
  int mg = r / 6, nt = r % 6;
  const bf16* A0 = win + (size_t)(mg * 64) * KINP;
  int n = lane & 15;
  const bf16* colp = xcT + (size_t)b * PP * KINP + (size_t)(nt * 16 + n) * KINP;
  v8f acc0 = {}, acc1 = {}, acc2 = {}, acc3 = {};
  for (int kt = 0; kt < KINP / 32; ++kt) {
    v16bf bb = frag_b_col(colp + kt * 32, lane, true);
    v16bf a0 = frag_a(A0 + (size_t)0  * KINP + kt * 32, KINP, lane);
    v16bf a1 = frag_a(A0 + (size_t)16 * KINP + kt * 32, KINP, lane);
    v16bf a2 = frag_a(A0 + (size_t)32 * KINP + kt * 32, KINP, lane);
    v16bf a3 = frag_a(A0 + (size_t)48 * KINP + kt * 32, KINP, lane);
    acc0 = wmma_bf16(a0, bb, acc0);
    acc1 = wmma_bf16(a1, bb, acc1);
    acc2 = wmma_bf16(a2, bb, acc2);
    acc3 = wmma_bf16(a3, bb, acc3);
  }
  int mb = (lane >> 4) << 3;
  int p  = nt * 16 + n;
  int m0 = mg * 64 + mb;
  bf16* out = qT + (size_t)b * PP * C_ + (size_t)p * C_ + m0;
  bool live = (p < P_);                               // zero pad rows >= 81
#pragma unroll
  for (int jj = 0; jj < 8; ++jj) out[jj]      = (bf16)(live ? acc0[jj] + b_in[m0 + jj]      : 0.f);
#pragma unroll
  for (int jj = 0; jj < 8; ++jj) out[16 + jj] = (bf16)(live ? acc1[jj] + b_in[m0 + 16 + jj] : 0.f);
#pragma unroll
  for (int jj = 0; jj < 8; ++jj) out[32 + jj] = (bf16)(live ? acc2[jj] + b_in[m0 + 32 + jj] : 0.f);
#pragma unroll
  for (int jj = 0; jj < 8; ++jj) out[48 + jj] = (bf16)(live ? acc3[jj] + b_in[m0 + 48 + jj] : 0.f);
}

// ---------------- K6: sim = q . k^T (f32 out) ----------------
// register-blocked 1 M-tile x 4 N-tiles : A fragment reused by 4 wmma
__global__ void k_gemm_sim(const bf16* __restrict__ qT, const bf16* __restrict__ kT,
                           float* __restrict__ sim) {
  int tile = blockIdx.x * 8 + (threadIdx.x >> 5);
  int lane = threadIdx.x & 31;
  if (tile >= B_ * 6 * 4) return;
  int b  = tile / 24;
  int r  = tile % 24;
  int mt = r / 4, ng = r % 4;                         // ng: group of 4 N tiles (j)
  const bf16* A = qT + (size_t)b * PP * C_ + (size_t)mt * 16 * C_;
  int n = lane & 15;
  const bf16* col0 = kT + (size_t)b * S_ * C_ + (size_t)(ng * 64 + n) * C_;
  v8f acc0 = {}, acc1 = {}, acc2 = {}, acc3 = {};
  for (int kt = 0; kt < C_ / 32; ++kt) {
    v16bf a  = frag_a(A + kt * 32, C_, lane);
    v16bf b0 = frag_b_col(col0 + (size_t)0  * C_ + kt * 32, lane, true);
    v16bf b1 = frag_b_col(col0 + (size_t)16 * C_ + kt * 32, lane, true);
    v16bf b2 = frag_b_col(col0 + (size_t)32 * C_ + kt * 32, lane, true);
    v16bf b3 = frag_b_col(col0 + (size_t)48 * C_ + kt * 32, lane, true);
    acc0 = wmma_bf16(a, b0, acc0);
    acc1 = wmma_bf16(a, b1, acc1);
    acc2 = wmma_bf16(a, b2, acc2);
    acc3 = wmma_bf16(a, b3, acc3);
  }
  int mb = (lane >> 4) << 3;
  float* out = sim + (size_t)b * PP * S_ + (size_t)(mt * 16 + mb) * S_ + ng * 64 + n;
#pragma unroll
  for (int jj = 0; jj < 8; ++jj) out[(size_t)jj * S_]      = acc0[jj];
#pragma unroll
  for (int jj = 0; jj < 8; ++jj) out[(size_t)jj * S_ + 16] = acc1[jj];
#pragma unroll
  for (int jj = 0; jj < 8; ++jj) out[(size_t)jj * S_ + 32] = acc2[jj];
#pragma unroll
  for (int jj = 0; jj < 8; ++jj) out[(size_t)jj * S_ + 48] = acc3[jj];
}

// ---------------- K7: softmax rows; emit attn (f32 -> d_out) and bf16 for xo ----------------
__global__ void k_softmax(const float* __restrict__ sim, bf16* __restrict__ attn,
                          float* __restrict__ attn_out) {
  __shared__ float red[256];
  int b = blockIdx.x / PP, p = blockIdx.x % PP;
  int j = threadIdx.x;
  float v = sim[(size_t)b * PP * S_ + (size_t)p * S_ + j];
  red[j] = v; __syncthreads();
  for (int o = 128; o > 0; o >>= 1) { if (j < o) red[j] = fmaxf(red[j], red[j + o]); __syncthreads(); }
  float mx = red[0]; __syncthreads();
  float e = __expf(v - mx);
  red[j] = e; __syncthreads();
  for (int o = 128; o > 0; o >>= 1) { if (j < o) red[j] += red[j + o]; __syncthreads(); }
  float a = e / red[0];
  attn[(size_t)b * PP * S_ + (size_t)p * S_ + j] = (bf16)a;
  if (p < P_) attn_out[(size_t)b * P_ * S_ + (size_t)p * S_ + j] = a;
}

// ---------------- K8: xo = attn . v -> xo[b][chan][p] f32 ----------------
// register-blocked 1 M-tile x 4 N-tiles : A fragment reused by 4 wmma
__global__ void k_gemm_xo(const bf16* __restrict__ attn, const bf16* __restrict__ vO,
                          float* __restrict__ xo) {
  int tile = blockIdx.x * 8 + (threadIdx.x >> 5);
  int lane = threadIdx.x & 31;
  if (tile >= B_ * 6 * MG) return;
  int b  = tile / (6 * MG);
  int r  = tile % (6 * MG);
  int mt = r / MG, ng = r % MG;                       // ng: group of 4 N tiles (o)
  const bf16* A = attn + (size_t)b * PP * S_ + (size_t)mt * 16 * S_;
  int n = lane & 15;
  const bf16* col0 = vO + (size_t)b * C_ * S_ + (size_t)(ng * 64 + n) * S_;
  v8f acc0 = {}, acc1 = {}, acc2 = {}, acc3 = {};
  for (int kt = 0; kt < S_ / 32; ++kt) {
    v16bf a  = frag_a(A + kt * 32, S_, lane);
    v16bf b0 = frag_b_col(col0 + (size_t)0  * S_ + kt * 32, lane, true);
    v16bf b1 = frag_b_col(col0 + (size_t)16 * S_ + kt * 32, lane, true);
    v16bf b2 = frag_b_col(col0 + (size_t)32 * S_ + kt * 32, lane, true);
    v16bf b3 = frag_b_col(col0 + (size_t)48 * S_ + kt * 32, lane, true);
    acc0 = wmma_bf16(a, b0, acc0);
    acc1 = wmma_bf16(a, b1, acc1);
    acc2 = wmma_bf16(a, b2, acc2);
    acc3 = wmma_bf16(a, b3, acc3);
  }
  int mb = (lane >> 4) << 3;
#pragma unroll
  for (int jj = 0; jj < 8; ++jj) {
    int p = mt * 16 + mb + jj;
    if (p < P_) {
      size_t base = (size_t)b * C_ * P_ + (size_t)p;
      xo[base + (size_t)(ng * 64 + n)      * P_] = acc0[jj];
      xo[base + (size_t)(ng * 64 + 16 + n) * P_] = acc1[jj];
      xo[base + (size_t)(ng * 64 + 32 + n) * P_] = acc2[jj];
      xo[base + (size_t)(ng * 64 + 48 + n) * P_] = acc3[jj];
    }
  }
}

// ---------------- K9: residual scatter-add (bandwidth bound) ----------------
__global__ void k_scatter(const float* __restrict__ gx, const float* __restrict__ bbox,
                          const float* __restrict__ xo, float* __restrict__ out) {
  size_t n = (size_t)blockIdx.x * 256 + threadIdx.x;
  int x = (int)(n & 63);
  int y = (int)((n >> 6) & 63);
  size_t bc = n >> 12;
  int c = (int)(bc % C_);
  int b = (int)(bc / C_);
  float v = gx[n];
  int bx1 = (int)(bbox[b * 4 + 0] * 64.f), by1 = (int)(bbox[b * 4 + 1] * 64.f);
  int bx2 = max((int)(bbox[b * 4 + 2] * 64.f), bx1 + 1);
  int by2 = max((int)(bbox[b * 4 + 3] * 64.f), by1 + 1);
  if (y >= by1 && y < by2 && x >= bx1 && x < bx2) {
    int sy = min(max((y - by1) * ROI / (by2 - by1), 0), ROI - 1);
    int sx = min(max((x - bx1) * ROI / (bx2 - bx1), 0), ROI - 1);
    v += xo[(size_t)b * C_ * P_ + (size_t)c * P_ + sy * ROI + sx];
  }
  out[n] = v;
}

// ---------------- launch ----------------
extern "C" void kernel_launch(void* const* d_in, const int* in_sizes, int n_in,
                              void* d_out, int out_size, void* d_ws, size_t ws_size,
                              hipStream_t stream) {
  (void)in_sizes; (void)n_in; (void)out_size; (void)ws_size;
  const float* global_x = (const float*)d_in[0];
  const float* context  = (const float*)d_in[1];
  const float* indic    = (const float*)d_in[2];
  const float* bbox     = (const float*)d_in[3];
  const float* ln_g     = (const float*)d_in[4];
  const float* ln_b     = (const float*)d_in[5];
  const float* cn_g     = (const float*)d_in[6];
  const float* cn_b     = (const float*)d_in[7];
  const float* w_in     = (const float*)d_in[8];
  const float* b_in     = (const float*)d_in[9];
  const float* w_ctx    = (const float*)d_in[10];
  const float* b_ctx    = (const float*)d_in[11];
  const float* w_out    = (const float*)d_in[12];
  const float* b_out    = (const float*)d_in[13];

  char* w = (char*)d_ws;
  bf16*  ctxT  = (bf16*)(w + OFF_CTXT);
  bf16*  kmapT = (bf16*)(w + OFF_KMAPT);
  bf16*  wctxb = (bf16*)(w + OFF_WCTX);
  bf16*  woutb = (bf16*)(w + OFF_WOUT);
  bf16*  winb  = (bf16*)(w + OFF_WIN);
  bf16*  kT    = (bf16*)(w + OFF_KT);
  bf16*  vO    = (bf16*)(w + OFF_VO);
  bf16*  xcT   = (bf16*)(w + OFF_XCT);
  bf16*  qT    = (bf16*)(w + OFF_QT);
  float* simf  = (float*)(w + OFF_SIM);
  bf16*  attnb = (bf16*)(w + OFF_ATT);
  float* xof   = (float*)(w + OFF_XO);

  float* outp    = (float*)d_out;
  float* attnout = outp + OUT0_ELEMS;

  // preprocessing (independent)
  k_ctx_norm<<<B_ * 32, 256, 0, stream>>>(context, cn_g, cn_b, ctxT, kmapT);
  k_pack_w<<<(C_*CTX + 9*C_*CTX + C_*KINP) / 256, 256, 0, stream>>>(w_ctx, w_out, w_in,
                                                                    wctxb, woutb, winb);
  k_roi_gn<<<B_ * 32, 256, 0, stream>>>(global_x, bbox, ln_g, ln_b, xcT);
  k_xcat_pad<<<B_, 256, 0, stream>>>(indic, xcT);

  // WMMA GEMMs (register-blocked 4-tile waves)
  k_gemm_k<<<(B_ * MG * 16) / 8, 256, 0, stream>>>(wctxb, kmapT, b_ctx, kT);
  k_conv_v<<<(B_ * MG * 16) / 8, 256, 0, stream>>>(woutb, ctxT, b_out, vO);
  k_gemm_q<<<(B_ * MG * 6) / 8, 256, 0, stream>>>(winb, xcT, b_in, qT);
  k_gemm_sim<<<(B_ * 6 * 4) / 8, 256, 0, stream>>>(qT, kT, simf);
  k_softmax<<<B_ * PP, 256, 0, stream>>>(simf, attnb, attnout);
  k_gemm_xo<<<(B_ * 6 * MG) / 8, 256, 0, stream>>>(attnb, vO, xof);

  // bandwidth-bound residual scatter
  k_scatter<<<(unsigned)(OUT0_ELEMS / 256), 256, 0, stream>>>(global_x, bbox, xof, outp);
}
